// MoEModel_90735479095898
// MI455X (gfx1250) — matrix operand order repaired and checked by reading, not compiled, not run on hardware
//
#include <hip/hip_runtime.h>

// ---------------------------------------------------------------------------
// MoE (16 experts, top-2) for MI455X / gfx1250
//   wave32 + v_wmma_f32_16x16x32_bf16 + global_load_async_to_lds_b128
// ---------------------------------------------------------------------------

typedef __attribute__((ext_vector_type(16))) __bf16 v16bf;
typedef __attribute__((ext_vector_type(8)))  float  v8f;

struct alignas(16) U4 { unsigned int x, y, z, w; };
struct alignas(16) F4 { float x, y, z, w; };
struct U8 { U4 lo, hi; };   // 32 bytes == v16bf

static_assert(sizeof(U8) == sizeof(v16bf), "frag size mismatch");

// Problem constants
#define BATCH 4096
#define DIN   1024
#define NEXP  16
#define TOPK  2
#define EMBD  256
#define H1D   2048
#define H2D   1024
#define OD    512
#define NSLOT (BATCH * TOPK)   // 8192

// Workspace layout (bytes)
#define OFF_COUNTS      0ull
#define OFF_CURSOR      64ull
#define OFF_OFFSETS     128ull
#define OFF_TOPI        256ull                            // 4096*2*4 = 32768
#define OFF_SLOT_OF     (OFF_TOPI + 32768ull)             // 32768
#define OFF_GATES       (OFF_SLOT_OF + 32768ull)          // 32768
#define OFF_SLOT_TOKEN  (OFF_GATES + 32768ull)            // 8192*4 = 32768
#define OFF_H1          (((OFF_SLOT_TOKEN + 32768ull) + 255ull) & ~255ull)
#define OFF_H2          (OFF_H1 + (unsigned long long)NSLOT * H1D * 2ull)
#define OFF_OUT3        OFF_H1   // alias: h1 dead before GEMM3 writes out3

// fp32 -> bf16 RNE, two values packed with a single v_perm_b32
__device__ __forceinline__ unsigned int f2bf_pack(float a, float b) {
    unsigned int ua = __builtin_bit_cast(unsigned int, a);
    unsigned int ub = __builtin_bit_cast(unsigned int, b);
    ua += 0x7fffu + ((ua >> 16) & 1u);
    ub += 0x7fffu + ((ub >> 16) & 1u);
    // result = { ub[31:16], ua[31:16] }  (low half from ua)
    return __builtin_amdgcn_perm(ub, ua, 0x07060302u);
}
__device__ __forceinline__ unsigned short f2bf1(float a) {
    unsigned int ua = __builtin_bit_cast(unsigned int, a);
    return (unsigned short)((ua + 0x7fffu + ((ua >> 16) & 1u)) >> 16);
}

// LDS byte-offset of a __shared__ pointer (addrspace(3) ptrtoint)
typedef __attribute__((address_space(3))) unsigned short lds_us_t;
__device__ __forceinline__ unsigned lds_off(const unsigned short* p) {
    return (unsigned)(unsigned long long)(lds_us_t*)(unsigned short*)(p);
}

// gfx1250 async copy: 16B global -> LDS, tracked by ASYNCcnt
__device__ __forceinline__ void async_b128(unsigned lds_byte_off, const void* g) {
    asm volatile("global_load_async_to_lds_b128 %0, %1, off"
                 :: "v"(lds_byte_off),
                    "v"((unsigned long long)(__SIZE_TYPE__)g)
                 : "memory");
}
__device__ __forceinline__ void wait_async0() {
    asm volatile("s_wait_asynccnt 0x0" ::: "memory");
}

// ---------------------------------------------------------------------------
// Unified WMMA GEMM:  D[M, ND] = act(A[M, KD] * B[KD, ND] + bias)
//   Block tile 128x128, K-step 32, 256 threads = 8 waves (4 x 2 wave grid),
//   each wave = 2x4 fragments of v_wmma_f32_16x16x32_bf16.
//   A_BF16: A streamed via double-buffered global_load_async_to_lds_b128.
//   else  : A fetched fp32 via register pipeline, converted to bf16.
//   B     : always fp32, fetched columnwise (coalesced), converted+transposed.
//   EXPERT: rows are compacted per-expert slots; A gathered via slot_token.
// ---------------------------------------------------------------------------
template <int KD, int ND, bool A_BF16, bool RELU, bool OUT_BF16, bool EXPERT>
__global__ __launch_bounds__(256) void moe_gemm(
    const float* __restrict__ A32, const unsigned short* __restrict__ A16,
    const float* __restrict__ Bw, const float* __restrict__ bias,
    float* __restrict__ out32, unsigned short* __restrict__ out16,
    const int* __restrict__ slot_token, const int* __restrict__ offsets,
    const int* __restrict__ counts)
{
    constexpr int KT   = KD / 32;
    constexpr int ABUF = A_BF16 ? 2 : 1;
    __shared__ unsigned short As[ABUF * 128 * 40];  // [buf][row][k], 80B row stride
    __shared__ unsigned short Bs[128 * 40];         // [n][k] transposed

    const int tid = threadIdx.x;
    int e = 0, mstart, mend;
    if constexpr (EXPERT) {
        e = (int)blockIdx.z;
        const int off = offsets[e];
        mstart = off + (int)blockIdx.y * 128;
        mend   = off + counts[e];
        if (mstart >= mend) return;            // early-exit past routed count
    } else {
        mstart = (int)blockIdx.y * 128;
        mend   = BATCH;
    }
    const int nb0 = (int)blockIdx.x * 128;
    const float* Bp    = Bw + (size_t)e * KD * ND;
    const float* biasp = bias + (EXPERT ? (size_t)e * ND : 0);

    // --- A-tile load assignment: thread -> (row, 16-wide k segment)
    const int ar  = tid >> 1;
    const int acs = (tid & 1) << 4;
    int arow = mstart + ar;
    if (arow >= mend) arow = mend - 1;         // clamp: no divergence, rows discarded
    const float*          a32src = nullptr;
    const unsigned short* a16src = nullptr;
    if constexpr (A_BF16) {
        a16src = A16 + (size_t)arow * KD + acs;
    } else {
        int tok;
        if constexpr (EXPERT) tok = slot_token[arow]; else tok = arow;
        a32src = A32 + (size_t)tok * KD + acs;
    }
    const unsigned aslot = (unsigned)(ar * 40 + acs) * 2u;  // byte offset in one A buf
    const unsigned asbase = lds_off(&As[0]);

    // --- B-tile load assignment: thread -> (n column, 16-deep k segment)
    const int bn  = tid & 127;
    const int bkh = tid >> 7;                  // 0 or 1
    const float* bsrc = Bp + (size_t)(bkh * 16) * ND + nb0 + bn;

    const int lane = tid & 31;
    const int wv   = tid >> 5;
    const int wm   = wv >> 1;                  // 0..3
    const int wn   = wv & 1;                   // 0..1
    const int g    = lane >> 4;                // lane half
    const int ml   = lane & 15;

    v8f acc[2][4];
    #pragma unroll
    for (int i = 0; i < 2; ++i)
        #pragma unroll
        for (int j = 0; j < 4; ++j)
            acc[i][j] = v8f{0.f, 0.f, 0.f, 0.f, 0.f, 0.f, 0.f, 0.f};

    // ---- pipeline prologue: start tile 0
    float bv[16];
    float av[16];
    {
        const float* bp = bsrc;
        #pragma unroll
        for (int j = 0; j < 16; ++j) bv[j] = bp[(size_t)j * ND];
        if constexpr (A_BF16) {
            async_b128(asbase + aslot,       a16src);
            async_b128(asbase + aslot + 16u, a16src + 8);
        } else {
            const F4* p = reinterpret_cast<const F4*>(a32src);
            #pragma unroll
            for (int j = 0; j < 4; ++j) {
                F4 f = p[j];
                av[4 * j + 0] = f.x; av[4 * j + 1] = f.y;
                av[4 * j + 2] = f.z; av[4 * j + 3] = f.w;
            }
        }
    }

    for (int kt = 0; kt < KT; ++kt) {
        const int cur = A_BF16 ? (kt & 1) : 0;

        // ---- store phase: regs -> LDS (B always; A only on fp32 path)
        if constexpr (!A_BF16) {
            U4 u0, u1;
            u0.x = f2bf_pack(av[0],  av[1]);  u0.y = f2bf_pack(av[2],  av[3]);
            u0.z = f2bf_pack(av[4],  av[5]);  u0.w = f2bf_pack(av[6],  av[7]);
            u1.x = f2bf_pack(av[8],  av[9]);  u1.y = f2bf_pack(av[10], av[11]);
            u1.z = f2bf_pack(av[12], av[13]); u1.w = f2bf_pack(av[14], av[15]);
            *reinterpret_cast<U4*>(&As[ar * 40 + acs])     = u0;
            *reinterpret_cast<U4*>(&As[ar * 40 + acs + 8]) = u1;
        }
        {
            U4 b0, b1;
            b0.x = f2bf_pack(bv[0],  bv[1]);  b0.y = f2bf_pack(bv[2],  bv[3]);
            b0.z = f2bf_pack(bv[4],  bv[5]);  b0.w = f2bf_pack(bv[6],  bv[7]);
            b1.x = f2bf_pack(bv[8],  bv[9]);  b1.y = f2bf_pack(bv[10], bv[11]);
            b1.z = f2bf_pack(bv[12], bv[13]); b1.w = f2bf_pack(bv[14], bv[15]);
            *reinterpret_cast<U4*>(&Bs[bn * 40 + bkh * 16])     = b0;
            *reinterpret_cast<U4*>(&Bs[bn * 40 + bkh * 16 + 8]) = b1;
        }
        if constexpr (A_BF16) wait_async0();   // A(kt) resident in As[cur]
        __syncthreads();

        // ---- issue next tile's loads; they overlap the WMMAs below
        if (kt + 1 < KT) {
            const float* bp = bsrc + (size_t)(kt + 1) * 32 * ND;
            #pragma unroll
            for (int j = 0; j < 16; ++j) bv[j] = bp[(size_t)j * ND];
            if constexpr (A_BF16) {
                const unsigned short* ga = a16src + (size_t)(kt + 1) * 32;
                const unsigned dst = asbase + (unsigned)(cur ^ 1) * 10240u + aslot;
                async_b128(dst,       ga);
                async_b128(dst + 16u, ga + 8);
            } else {
                const F4* p = reinterpret_cast<const F4*>(a32src + (size_t)(kt + 1) * 32);
                #pragma unroll
                for (int j = 0; j < 4; ++j) {
                    F4 f = p[j];
                    av[4 * j + 0] = f.x; av[4 * j + 1] = f.y;
                    av[4 * j + 2] = f.z; av[4 * j + 3] = f.w;
                }
            }
        }

        // ---- build fragments per ISA 16-bit layouts, then 8 WMMAs
        const unsigned short* Ab = &As[cur * 5120];
        v16bf afr[2], bfr[4];
        #pragma unroll
        for (int mf = 0; mf < 2; ++mf) {
            const int m = wm * 32 + mf * 16 + ml;
            U8 fr;
            fr.lo = *reinterpret_cast<const U4*>(&Ab[m * 40 + g * 8]);       // K 8g..8g+7
            fr.hi = *reinterpret_cast<const U4*>(&Ab[m * 40 + 16 + g * 8]);  // K 16+8g..
            afr[mf] = __builtin_bit_cast(v16bf, fr);
        }
        #pragma unroll
        for (int nf = 0; nf < 4; ++nf) {
            const int n = wn * 64 + nf * 16 + ml;
            U8 fr;
            fr.lo = *reinterpret_cast<const U4*>(&Bs[n * 40 + g * 8]);       // K 8g..8g+7
            fr.hi = *reinterpret_cast<const U4*>(&Bs[n * 40 + 16 + g * 8]);  // K 16+8g..
            bfr[nf] = __builtin_bit_cast(v16bf, fr);
        }
        #pragma unroll
        for (int mf = 0; mf < 2; ++mf)
            #pragma unroll
            for (int nf = 0; nf < 4; ++nf)
                acc[mf][nf] = __builtin_amdgcn_wmma_f32_16x16x32_bf16(
                    false, afr[mf], false, bfr[nf], (short)0, acc[mf][nf],
                    false, false);
        __syncthreads();
    }

    // ---- epilogue: bias (+ReLU), store fp32 or bf16, guard past-count rows
    #pragma unroll
    for (int mf = 0; mf < 2; ++mf) {
        #pragma unroll
        for (int nf = 0; nf < 4; ++nf) {
            const int col = nb0 + wn * 64 + nf * 16 + ml;
            const float bvs = biasp[col];
            #pragma unroll
            for (int r = 0; r < 8; ++r) {
                const int row = mstart + wm * 32 + mf * 16 + g * 8 + r;
                float v = acc[mf][nf][r] + bvs;
                if constexpr (RELU) v = v > 0.f ? v : 0.f;
                if (row < mend) {
                    if constexpr (OUT_BF16)
                        out16[(size_t)row * ND + col] = f2bf1(v);
                    else
                        out32[(size_t)row * ND + col] = v;
                }
            }
        }
    }
}

// ---------------------------------------------------------------------------
// Router scores, softmax (gating_probs output), top-2 gates, expert counts
// ---------------------------------------------------------------------------
__global__ __launch_bounds__(256) void score_topk_kernel(
    const float* __restrict__ q, const float* __restrict__ emb,
    float* __restrict__ gp, int* __restrict__ topi, float* __restrict__ gates,
    int* __restrict__ counts)
{
    __shared__ float se[NEXP * EMBD];
    const int tid = threadIdx.x;
    for (int i = tid; i < NEXP * EMBD; i += 256) se[i] = emb[i];
    __syncthreads();

    const int b = (int)blockIdx.x * 256 + tid;
    const float* qr = q + (size_t)b * EMBD;

    float acc[NEXP];
    #pragma unroll
    for (int e = 0; e < NEXP; ++e) acc[e] = 0.f;
    for (int j = 0; j < EMBD; ++j) {
        const float qj = qr[j];
        #pragma unroll
        for (int e = 0; e < NEXP; ++e) {
            const float d = qj - se[e * EMBD + j];
            acc[e] += d * d;
        }
    }
    // scores = -d2 ; softmax over 16
    float mx = -acc[0];
    #pragma unroll
    for (int e = 1; e < NEXP; ++e) mx = fmaxf(mx, -acc[e]);
    float s = 0.f, ex[NEXP];
    #pragma unroll
    for (int e = 0; e < NEXP; ++e) { ex[e] = __expf(-acc[e] - mx); s += ex[e]; }
    const float inv = 1.f / s;
    #pragma unroll
    for (int e = 0; e < NEXP; ++e) gp[(size_t)b * NEXP + e] = ex[e] * inv;

    // top-2 (first occurrence on ties, matching lax.top_k)
    int i1 = 0; float v1 = -acc[0];
    #pragma unroll
    for (int e = 1; e < NEXP; ++e)
        if (-acc[e] > v1) { v1 = -acc[e]; i1 = e; }
    int i2 = (i1 == 0) ? 1 : 0; float v2 = -acc[i2];
    #pragma unroll
    for (int e = 0; e < NEXP; ++e)
        if (e != i1 && -acc[e] > v2) { v2 = -acc[e]; i2 = e; }

    float g2 = __expf(v2 - v1);
    const float den = 1.f + g2;
    const float g1 = 1.f / den;
    g2 = g2 / den;

    topi[2 * b] = i1; topi[2 * b + 1] = i2;
    gates[2 * b] = g1; gates[2 * b + 1] = g2;
    atomicAdd(&counts[i1], 1);
    atomicAdd(&counts[i2], 1);
}

__global__ void zero_meta_kernel(int* counts) {
    if (threadIdx.x < NEXP) counts[threadIdx.x] = 0;
}

__global__ void scan_kernel(const int* __restrict__ counts,
                            int* __restrict__ offsets, int* __restrict__ cursor) {
    if (threadIdx.x == 0) {
        int s = 0;
        for (int e = 0; e < NEXP; ++e) {
            offsets[e] = s; cursor[e] = s; s += counts[e];
        }
    }
}

__global__ __launch_bounds__(256) void fill_slots_kernel(
    const int* __restrict__ topi, int* __restrict__ cursor,
    int* __restrict__ slot_token, int* __restrict__ slot_of)
{
    const int b = (int)blockIdx.x * 256 + threadIdx.x;
    #pragma unroll
    for (int k = 0; k < TOPK; ++k) {
        const int e = topi[2 * b + k];
        const int s = atomicAdd(&cursor[e], 1);
        slot_token[s] = b;
        slot_of[2 * b + k] = s;
    }
}

__global__ __launch_bounds__(256) void combine_kernel(
    const float* __restrict__ out3, const int* __restrict__ slot_of,
    const float* __restrict__ gates, float* __restrict__ fin)
{
    const int b = (int)blockIdx.x;
    const int s0 = slot_of[2 * b], s1 = slot_of[2 * b + 1];
    const float g0 = gates[2 * b], g1 = gates[2 * b + 1];
    const float* r0 = out3 + (size_t)s0 * OD;
    const float* r1 = out3 + (size_t)s1 * OD;
    for (int o = (int)threadIdx.x; o < OD; o += 256)
        fin[(size_t)b * OD + o] = g0 * r0[o] + g1 * r1[o];
}

// ---------------------------------------------------------------------------
extern "C" void kernel_launch(void* const* d_in, const int* in_sizes, int n_in,
                              void* d_out, int out_size, void* d_ws, size_t ws_size,
                              hipStream_t stream)
{
    const float* x   = (const float*)d_in[0];
    const float* Wr  = (const float*)d_in[1];
    const float* br  = (const float*)d_in[2];
    const float* emb = (const float*)d_in[3];
    const float* W1  = (const float*)d_in[4];
    const float* b1  = (const float*)d_in[5];
    const float* W2  = (const float*)d_in[6];
    const float* b2  = (const float*)d_in[7];
    const float* W3  = (const float*)d_in[8];
    const float* b3  = (const float*)d_in[9];

    float* fin  = (float*)d_out;                         // [4096, 512]
    float* qout = fin + (size_t)BATCH * OD;              // [4096, 256]
    float* gpout = qout + (size_t)BATCH * EMBD;          // [4096, 16]

    char* w = (char*)d_ws;
    int* counts     = (int*)(w + OFF_COUNTS);
    int* cursor     = (int*)(w + OFF_CURSOR);
    int* offsets    = (int*)(w + OFF_OFFSETS);
    int* topi       = (int*)(w + OFF_TOPI);
    int* slot_of    = (int*)(w + OFF_SLOT_OF);
    float* gates    = (float*)(w + OFF_GATES);
    int* slot_token = (int*)(w + OFF_SLOT_TOKEN);
    unsigned short* h1 = (unsigned short*)(w + OFF_H1);
    unsigned short* h2 = (unsigned short*)(w + OFF_H2);
    float* out3        = (float*)(w + OFF_OUT3);         // aliases h1 (dead)

    zero_meta_kernel<<<1, 32, 0, stream>>>(counts);

    // Router: q = x @ Wr + br   [4096,1024]x[1024,256]
    moe_gemm<DIN, EMBD, false, false, false, false>
        <<<dim3(EMBD / 128, BATCH / 128, 1), 256, 0, stream>>>(
            x, nullptr, Wr, br, qout, nullptr, nullptr, nullptr, nullptr);

    score_topk_kernel<<<BATCH / 256, 256, 0, stream>>>(
        qout, emb, gpout, topi, gates, counts);

    scan_kernel<<<1, 32, 0, stream>>>(counts, offsets, cursor);

    fill_slots_kernel<<<BATCH / 256, 256, 0, stream>>>(
        topi, cursor, slot_token, slot_of);

    // Expert layer 1: h1 = relu(gather(x) @ W1[e] + b1[e])  -> bf16
    moe_gemm<DIN, H1D, false, true, true, true>
        <<<dim3(H1D / 128, BATCH / 128, NEXP), 256, 0, stream>>>(
            x, nullptr, W1, b1, nullptr, h1, slot_token, offsets, counts);

    // Expert layer 2: h2 = relu(h1 @ W2[e] + b2[e]) -> bf16  (async A tiles)
    moe_gemm<H1D, H2D, true, true, true, true>
        <<<dim3(H2D / 128, BATCH / 128, NEXP), 256, 0, stream>>>(
            nullptr, h1, W2, b2, nullptr, h2, nullptr, offsets, counts);

    // Expert layer 3: out3 = h2 @ W3[e] + b3[e] -> fp32     (async A tiles)
    moe_gemm<H2D, OD, true, false, false, true>
        <<<dim3(OD / 128, BATCH / 128, NEXP), 256, 0, stream>>>(
            nullptr, h2, W3, b3, out3, nullptr, nullptr, offsets, counts);

    // final[b] = g0*out3[slot0] + g1*out3[slot1]
    combine_kernel<<<BATCH, 256, 0, stream>>>(out3, slot_of, gates, fin);
}
